// glassDBN_68169720922297
// MI455X (gfx1250) — matrix operand order repaired and checked by reading, not compiled, not run on hardware
//
#include <hip/hip_runtime.h>
#include <hip/hip_bf16.h>

// ---------------------------------------------------------------------------
// CDNA5 (gfx1250) implementation. bf16 WMMA (v_wmma_f32_16x16x32_bf16) for all
// GEMMs, fp32 accumulation. wave32: wave = tid>>5, lane = tid&31.
// ---------------------------------------------------------------------------

typedef __attribute__((ext_vector_type(16))) __bf16 v16bf;
typedef __attribute__((ext_vector_type(8)))  float  v8f;

union FragV { v16bf v; uint4 u[2]; };

__device__ __forceinline__ v8f zero8() {
  v8f z = {0.f,0.f,0.f,0.f,0.f,0.f,0.f,0.f};
  return z;
}
__device__ __forceinline__ float sigmoidf_(float x) { return 1.f / (1.f + __expf(-x)); }

// One 16x16 D tile of  D = A(MxK) * B(NxK)^T  (+bias) using 16x16x32 bf16 WMMA.
// A: row-major MxK (LDS, 16B-aligned rows), B: row-major NxK (weights natural
// (out,in) layout -> lane n reads two 16-byte chunks). K % 32 == 0.
// A-fragment layout (16-bit A, wave32): lane L: M=L%16, half=L/16,
//   elems 0..7  <- K = k0 + half*8 .. +7
//   elems 8..15 <- K = k0 + 16 + half*8 .. +7
// B-fragment: lane L: N=L%16, same K mapping as the A-fragment.
// D (f32 16x16): lane L holds column N=L%16, rows M = vg + 8*half.
// Compile-time flags avoid runtime-null branches in the epilogue.
template <bool HAS_BIAS, bool RELU, bool OUT_BF16, bool OUT_F32>
__device__ __forceinline__ void gemm_tiles(
    const __bf16* __restrict__ A, int lda,
    const __bf16* __restrict__ B, int ldb,
    const float*  __restrict__ bias,
    __bf16* __restrict__ D, int ldd,          // bf16 out (if OUT_BF16)
    float*  __restrict__ Df, int lddf,        // f32 out (if OUT_F32)
    int Mtiles, int Ntiles, int K,
    int wave, int nwaves, int lane)
{
  const int l15  = lane & 15;
  const int half = lane >> 4;
  for (int s = wave; s < Mtiles * Ntiles; s += nwaves) {
    const int mt = s / Ntiles, nt = s % Ntiles;
    const int m0 = mt * 16, n0 = nt * 16;
    const int mrow = m0 + l15;
    const int ncol = n0 + l15;
    v8f acc = zero8();
    for (int k0 = 0; k0 < K; k0 += 32) {
      FragV a, b;
      const __bf16* pa = A + mrow * lda + k0 + half * 8;
      a.u[0] = *(const uint4*)(pa);
      a.u[1] = *(const uint4*)(pa + 16);
      const __bf16* pb = B + (size_t)ncol * ldb + k0 + half * 8;
      b.u[0] = *(const uint4*)(pb);
      b.u[1] = *(const uint4*)(pb + 16);
      acc = __builtin_amdgcn_wmma_f32_16x16x32_bf16(
          false, a.v, false, b.v, (short)0, acc, false, false);
    }
    float bv = 0.f;
    if (HAS_BIAS) bv = bias[ncol];
#pragma unroll
    for (int vg = 0; vg < 8; ++vg) {
      float v = acc[vg] + bv;
      if (RELU) v = fmaxf(v, 0.f);
      const int row = m0 + vg + 8 * half;
      if (OUT_BF16) D [(size_t)row * ldd  + ncol] = (__bf16)v;
      if (OUT_F32)  Df[(size_t)row * lddf + ncol] = v;
    }
  }
}

// ---------------------------------------------------------------------------
// Weight fp32 -> bf16 convert with zero padding.
// ---------------------------------------------------------------------------
__global__ void cvt_kernel(const float* __restrict__ src, __bf16* __restrict__ dst,
                           int rows, int cols, int src_rows, int src_cols)
{
  int i = blockIdx.x * 256 + threadIdx.x;
  if (i < rows * cols) {
    int r = i / cols, c = i % cols;
    float v = (r < src_rows && c < src_cols) ? src[(size_t)r * src_cols + c] : 0.f;
    dst[i] = (__bf16)v;
  }
}

// ---------------------------------------------------------------------------
// Recurrent scan: one block per batch element b. 256 threads = 8 wave32s.
// Dynamic LDS pool (~185 KB, WGP has 320 KB).
// ---------------------------------------------------------------------------
__global__ void __launch_bounds__(256) scan_kernel(
    const float* __restrict__ x,        // (64,256,53)
    const float* __restrict__ w_emb,    // (64,1)
    const float* __restrict__ b_emb,    // (64)
    const __bf16* __restrict__ wih, const __bf16* __restrict__ whh,
    const float* __restrict__ bih, const float* __restrict__ bhh,
    const __bf16* __restrict__ qw0, const float* __restrict__ qb0,
    const __bf16* __restrict__ qw1, const float* __restrict__ qb1,
    const __bf16* __restrict__ qw2, const float* __restrict__ qb2,
    const __bf16* __restrict__ kw0, const float* __restrict__ kb0,
    const __bf16* __restrict__ kw1, const float* __restrict__ kb1,
    const __bf16* __restrict__ kw2, const float* __restrict__ kb2,
    const float* __restrict__ gate_bias,   // (53,53)
    const float* __restrict__ pred_w,      // (64)
    const float* __restrict__ pred_b,      // (1)
    float* __restrict__ mix_out,           // (64,256,53,53)
    float* __restrict__ pred_out)          // (64,255,53)
{
  extern __shared__ char smem[];
  __bf16* sE   = (__bf16*)(smem);             // 64x64 embedded       8192
  __bf16* sH   = (__bf16*)(smem + 8192);      // 64x64 h prev (bf16)  8192
  float*  sHf  = (float*) (smem + 16384);     // 64x64 h prev (f32)  16384
  __bf16* sGI  = (__bf16*)(smem + 32768);     // 64x192 gi           24576
  __bf16* sGH  = (__bf16*)(smem + 57344);     // 64x192 gh           24576
  __bf16* sHG  = (__bf16*)(smem + 81920);     // 64x64 h_gru          8192
  __bf16* sHGT = (__bf16*)(smem + 90112);     // 64x64 h_gru^T        8192
  __bf16* sA0  = (__bf16*)(smem + 98304);     // 64x128 q ping       16384
  __bf16* sB0  = (__bf16*)(smem + 114688);    // 64x128 k ping       16384
  __bf16* sA1  = (__bf16*)(smem + 131072);    // 64x128 q pong       16384
  __bf16* sB1  = (__bf16*)(smem + 147456);    // 64x128 k pong       16384
  float*  sTR  = (float*) (smem + 163840);    // 64x64 transfer f32  16384
  __bf16* sTRB = (__bf16*)(smem + 180224);    // 64x64 gated bf16     8192
  float*  sRed = (float*) (smem + 188416);    // 256 reduce           1024

  const int b    = blockIdx.x;
  const int tid  = threadIdx.x;
  const int wave = tid >> 5;
  const int lane = tid & 31;

  for (int i = tid; i < 64 * 64; i += 256) { sH[i] = (__bf16)0.f; sHf[i] = 0.f; }
  __syncthreads();

  for (int t = 0; t < 256; ++t) {
    // ---- A) embedded: e[c][j] = x[b,t,c]*w_emb[j] + b_emb[j], pad rows -> 0
    const float* xb = x + ((size_t)b * 256 + t) * 53;
    for (int i = tid; i < 64 * 64; i += 256) {
      int c = i >> 6, j = i & 63;
      float v = (c < 53) ? (xb[c] * w_emb[j] + b_emb[j]) : 0.f;
      sE[i] = (__bf16)v;
    }
    __syncthreads();

    // ---- B) gi = e @ w_ih^T + b_ih ; gh = h @ w_hh^T + b_hh  (64x192, K=64)
    gemm_tiles<true,false,true,false>(sE, 64, wih, 64, bih, sGI, 192, nullptr, 0,
                                      4, 12, 64, wave, 8, lane);
    gemm_tiles<true,false,true,false>(sH, 64, whh, 64, bhh, sGH, 192, nullptr, 0,
                                      4, 12, 64, wave, 8, lane);
    __syncthreads();

    // ---- C) GRU elementwise -> h_gru (and transposed copy)
    for (int i = tid; i < 64 * 64; i += 256) {
      int c = i >> 6, h = i & 63;
      float ir = (float)sGI[c * 192 + h];
      float iz = (float)sGI[c * 192 + 64 + h];
      float in = (float)sGI[c * 192 + 128 + h];
      float hr = (float)sGH[c * 192 + h];
      float hz = (float)sGH[c * 192 + 64 + h];
      float hn = (float)sGH[c * 192 + 128 + h];
      float r = sigmoidf_(ir + hr);
      float z = sigmoidf_(iz + hz);
      float n = tanhf(in + r * hn);
      float hg = (1.f - z) * n + z * sHf[i];
      sHG[i] = (__bf16)hg;
      sHGT[h * 64 + c] = (__bf16)hg;
    }
    __syncthreads();

    // ---- D) q/k 3-layer MLPs (64x128 each layer)
    gemm_tiles<true,true,true,false>(sHG, 64, qw0, 64, qb0, sA0, 128, nullptr, 0,
                                     4, 8, 64, wave, 8, lane);
    gemm_tiles<true,true,true,false>(sHG, 64, kw0, 64, kb0, sB0, 128, nullptr, 0,
                                     4, 8, 64, wave, 8, lane);
    __syncthreads();
    gemm_tiles<true,true,true,false>(sA0, 128, qw1, 128, qb1, sA1, 128, nullptr, 0,
                                     4, 8, 128, wave, 8, lane);
    gemm_tiles<true,true,true,false>(sB0, 128, kw1, 128, kb1, sB1, 128, nullptr, 0,
                                     4, 8, 128, wave, 8, lane);
    __syncthreads();
    gemm_tiles<true,false,true,false>(sA1, 128, qw2, 128, qb2, sA0, 128, nullptr, 0,
                                      4, 8, 128, wave, 8, lane);
    gemm_tiles<true,false,true,false>(sB1, 128, kw2, 128, kb2, sB0, 128, nullptr, 0,
                                      4, 8, 128, wave, 8, lane);
    __syncthreads();

    // ---- E) transfer = q @ k^T (k row-major (d,a) is exactly the B layout)
    gemm_tiles<false,false,false,true>(sA0, 128, sB0, 128, nullptr, nullptr, 0,
                                       sTR, 64, 4, 4, 128, wave, 8, lane);
    __syncthreads();

    // Frobenius norm over valid 53x53 (two-stage reduction)
    float part = 0.f;
    for (int i = tid; i < 53 * 53; i += 256) {
      int c = i / 53, d = i % 53;
      float v = sTR[c * 64 + d];
      part += v * v;
    }
    sRed[tid] = part;
    __syncthreads();
    if (tid < 32) {
      float s = sRed[tid];
#pragma unroll
      for (int j = 1; j < 8; ++j) s += sRed[tid + j * 32];
      sRed[tid] = s;
    }
    __syncthreads();
    if (tid == 0) {
      float s = 0.f;
#pragma unroll
      for (int j = 0; j < 32; ++j) s += sRed[j];
      sRed[0] = 1.f / sqrtf(s);
    }
    __syncthreads();

    // ---- F) normalize + gate; emit mix; pad -> 0 so h_new K-contraction is exact
    {
      const float inv = sRed[0];
      for (int i = tid; i < 64 * 64; i += 256) {
        int c = i >> 6, d = i & 63;
        float v = 0.f;
        if (c < 53 && d < 53) {
          float tr = sTR[i] * inv;
          float g  = sigmoidf_(fabsf(tr) + gate_bias[c * 53 + d]);
          v = tr * g;
          mix_out[((size_t)(b * 256 + t)) * 2809 + c * 53 + d] = v;
        }
        sTRB[i] = (__bf16)v;
      }
    }
    __syncthreads();

    // ---- G) h_new = transfer @ h_gru   (B fragment = h_gru^T rows)
    gemm_tiles<false,false,true,true>(sTRB, 64, sHGT, 64, nullptr, sH, 64,
                                      sHf, 64, 4, 4, 64, wave, 8, lane);
    __syncthreads();

    // ---- predicted (uses h at steps 0..T-2)
    if (t < 255) {
      for (int c = tid; c < 53; c += 256) {
        float acc = pred_b[0];
#pragma unroll
        for (int h = 0; h < 64; ++h) acc += sHf[c * 64 + h] * pred_w[h];
        pred_out[((size_t)b * 255 + t) * 53 + c] = acc;
      }
    }
    __syncthreads();
  }
}

// ---------------------------------------------------------------------------
// Classifier GEMM: Dst(MxN,bf16) = relu(A(MxKact) @ B(NxK)^T + bias).
// Block tile 64 x BN (BN = 64 or 128). A staged through LDS: fp32 path uses
// VALU convert staging; bf16 path uses GLOBAL_LOAD_ASYNC_TO_LDS_B128 (one
// 16-byte async transfer per lane, ASYNCcnt-tracked, s_wait_asynccnt before
// the barrier). K % 32 == 0, M % 64 == 0, N % BN == 0.
// ---------------------------------------------------------------------------
__global__ void __launch_bounds__(256) clf_gemm_kernel(
    const void* __restrict__ Aptr, int a_is_f32, int Kact,
    const __bf16* __restrict__ B, const float* __restrict__ bias, int bias_n,
    __bf16* __restrict__ Dst, int N, int K, int BN)
{
  __shared__ __bf16 sA[64 * 32];
  const int m0 = blockIdx.x * 64, n0 = blockIdx.y * BN;
  const int tid = threadIdx.x, wave = tid >> 5, lane = tid & 31;
  const int l15 = lane & 15, half = lane >> 4;
  const int Ntiles = BN >> 4;
  const int nsub = (4 * Ntiles) >> 3;     // subtiles per wave (<= 4)

  // Per-thread async-staging geometry: thread tid owns row tid>>2, 8 bf16
  // starting at column (tid&3)*8  (64 rows x 32 cols == 256 x 16B).
  const int st_r = tid >> 2;
  const int st_c = (tid & 3) * 8;
  const unsigned lds_off =
      (unsigned)(uintptr_t)(&sA[0]) + (unsigned)tid * 16u;  // low 32b = LDS offset

  v8f acc[4];
#pragma unroll
  for (int s = 0; s < 4; ++s) acc[s] = zero8();

  for (int k0 = 0; k0 < K; k0 += 32) {
    if (a_is_f32) {
      const float* Af = (const float*)Aptr;
      for (int i = tid; i < 64 * 32; i += 256) {
        int r = i >> 5, c = i & 31;
        int kk = k0 + c;
        float v = (kk < Kact) ? Af[(size_t)(m0 + r) * Kact + kk] : 0.f;
        sA[i] = (__bf16)v;
      }
    } else {
      const __bf16* Abf = (const __bf16*)Aptr;
      unsigned long long ga =
          (unsigned long long)(uintptr_t)(Abf + (size_t)(m0 + st_r) * K + k0 + st_c);
      asm volatile("global_load_async_to_lds_b128 %0, %1, off"
                   :: "v"(lds_off), "v"(ga) : "memory");
      asm volatile("s_wait_asynccnt 0x0" ::: "memory");
    }
    __syncthreads();
    for (int s = 0; s < nsub; ++s) {
      int sub = wave + s * 8;
      int mt = sub & 3, nt = sub >> 2;
      FragV a, bfr;
      const __bf16* pa = sA + (mt * 16 + l15) * 32 + half * 8;
      a.u[0] = *(const uint4*)(pa);
      a.u[1] = *(const uint4*)(pa + 16);
      const __bf16* pb = B + (size_t)(n0 + nt * 16 + l15) * K + k0 + half * 8;
      __builtin_prefetch((const void*)(pb + 32), 0, 1);
      bfr.u[0] = *(const uint4*)(pb);
      bfr.u[1] = *(const uint4*)(pb + 16);
      acc[s] = __builtin_amdgcn_wmma_f32_16x16x32_bf16(
          false, a.v, false, bfr.v, (short)0, acc[s], false, false);
    }
    __syncthreads();
  }
  for (int s = 0; s < nsub; ++s) {
    int sub = wave + s * 8;
    int mt = sub & 3, nt = sub >> 2;
    int ncol = n0 + nt * 16 + l15;
    float bv = (ncol < bias_n) ? bias[ncol] : 0.f;
#pragma unroll
    for (int vg = 0; vg < 8; ++vg) {
      float v = fmaxf(acc[s][vg] + bv, 0.f);
      int row = m0 + mt * 16 + vg + 8 * half;
      Dst[(size_t)row * N + ncol] = (__bf16)v;
    }
  }
}

// logits[b] = mean_t(z2[b,t] @ w2^T) + b2
__global__ void __launch_bounds__(256) logits_kernel(
    const __bf16* __restrict__ z2, const float* __restrict__ w2,
    const float* __restrict__ b2, float* __restrict__ logits)
{
  __shared__ float mz[704];
  const int b = blockIdx.x, tid = threadIdx.x;
  for (int j = tid; j < 704; j += 256) {
    float s = 0.f;
    for (int t = 0; t < 256; ++t) s += (float)z2[(size_t)(b * 256 + t) * 704 + j];
    mz[j] = s * (1.f / 256.f);
  }
  __syncthreads();
  if (tid < 2) {
    float s = b2[tid];
    for (int j = 0; j < 702; ++j) s += mz[j] * w2[tid * 702 + j];
    logits[b * 2 + tid] = s;
  }
}

__global__ void meanfnc_kernel(const float* __restrict__ mix, float* __restrict__ out)
{
  const int b = blockIdx.x;
  const int p = blockIdx.y * 256 + threadIdx.x;
  if (p < 2809) {
    float s = 0.f;
    for (int t = 0; t < 256; ++t) s += mix[((size_t)(b * 256 + t)) * 2809 + p];
    out[(size_t)b * 2809 + p] = s * (1.f / 256.f);
  }
}

__global__ void xcopy_kernel(const float* __restrict__ x, float* __restrict__ xo)
{
  int i = blockIdx.x * 256 + threadIdx.x;
  if (i < 64 * 255 * 53) {
    int b = i / (255 * 53);
    int r = i % (255 * 53);
    int t = r / 53, c = r % 53;
    xo[i] = x[((size_t)b * 256 + t + 1) * 53 + c];
  }
}

// ---------------------------------------------------------------------------
// Host orchestration
// ---------------------------------------------------------------------------
// Workspace layout (bytes)
static constexpr size_t OFF_WIH = 0;                         // 192x64 bf16
static constexpr size_t OFF_WHH = 24576;                     // 192x64
static constexpr size_t OFF_QW0 = 49152;                     // 128x64
static constexpr size_t OFF_QW1 = 65536;                     // 128x128
static constexpr size_t OFF_QW2 = 98304;                     // 128x128
static constexpr size_t OFF_KW0 = 131072;                    // 128x64
static constexpr size_t OFF_KW1 = 147456;                    // 128x128
static constexpr size_t OFF_KW2 = 180224;                    // 128x128
static constexpr size_t OFF_CW0 = 212992;                    // 1408x2816 (padded)
static constexpr size_t OFF_CW1 = 8142848;                   // 704x1408 (padded)
static constexpr size_t OFF_Z1  = 10147840;                  // 16384x1408 bf16
static constexpr size_t OFF_Z2  = 56285184;                  // 16384x704 bf16

// Output layout (fp32 elements, concatenated in return order)
static constexpr size_t OUT_LOGITS = 0;                      // 64*2
static constexpr size_t OUT_MEAN   = 128;                    // 64*2809
static constexpr size_t OUT_MIX    = 179904;                 // 64*256*2809
static constexpr size_t OUT_PRED   = 46202560;               // 64*255*53
static constexpr size_t OUT_X      = 47067520;               // 64*255*53

static constexpr size_t SMEM_SCAN = 189440;

extern "C" void kernel_launch(void* const* d_in, const int* in_sizes, int n_in,
                              void* d_out, int out_size, void* d_ws, size_t ws_size,
                              hipStream_t stream)
{
  const float* x       = (const float*)d_in[0];
  const float* w_emb   = (const float*)d_in[1];
  const float* b_emb   = (const float*)d_in[2];
  const float* w_ih    = (const float*)d_in[3];
  const float* w_hh    = (const float*)d_in[4];
  const float* b_ih    = (const float*)d_in[5];
  const float* b_hh    = (const float*)d_in[6];
  const float* q_w0    = (const float*)d_in[7];
  const float* q_b0    = (const float*)d_in[8];
  const float* q_w1    = (const float*)d_in[9];
  const float* q_b1    = (const float*)d_in[10];
  const float* q_w2    = (const float*)d_in[11];
  const float* q_b2    = (const float*)d_in[12];
  const float* k_w0    = (const float*)d_in[13];
  const float* k_b0    = (const float*)d_in[14];
  const float* k_w1    = (const float*)d_in[15];
  const float* k_b1    = (const float*)d_in[16];
  const float* k_w2    = (const float*)d_in[17];
  const float* k_b2    = (const float*)d_in[18];
  const float* gate_b  = (const float*)d_in[19];
  const float* pred_w  = (const float*)d_in[20];
  const float* pred_b  = (const float*)d_in[21];
  const float* clf_w0  = (const float*)d_in[22];
  const float* clf_b0  = (const float*)d_in[23];
  const float* clf_w1  = (const float*)d_in[24];
  const float* clf_b1  = (const float*)d_in[25];
  const float* clf_w2  = (const float*)d_in[26];
  const float* clf_b2  = (const float*)d_in[27];

  char* ws = (char*)d_ws;
  __bf16* wih = (__bf16*)(ws + OFF_WIH);
  __bf16* whh = (__bf16*)(ws + OFF_WHH);
  __bf16* qw0 = (__bf16*)(ws + OFF_QW0);
  __bf16* qw1 = (__bf16*)(ws + OFF_QW1);
  __bf16* qw2 = (__bf16*)(ws + OFF_QW2);
  __bf16* kw0 = (__bf16*)(ws + OFF_KW0);
  __bf16* kw1 = (__bf16*)(ws + OFF_KW1);
  __bf16* kw2 = (__bf16*)(ws + OFF_KW2);
  __bf16* cw0 = (__bf16*)(ws + OFF_CW0);
  __bf16* cw1 = (__bf16*)(ws + OFF_CW1);
  __bf16* z1  = (__bf16*)(ws + OFF_Z1);
  __bf16* z2  = (__bf16*)(ws + OFF_Z2);

  float* out    = (float*)d_out;
  float* logits = out + OUT_LOGITS;
  float* meanf  = out + OUT_MEAN;
  float* mix    = out + OUT_MIX;
  float* pred   = out + OUT_PRED;
  float* xout   = out + OUT_X;

  auto cvt = [&](const float* s, __bf16* d, int R, int C, int sr, int sc) {
    int n = R * C;
    cvt_kernel<<<(n + 255) / 256, 256, 0, stream>>>(s, d, R, C, sr, sc);
  };
  cvt(w_ih, wih, 192, 64, 192, 64);
  cvt(w_hh, whh, 192, 64, 192, 64);
  cvt(q_w0, qw0, 128, 64, 128, 64);
  cvt(q_w1, qw1, 128, 128, 128, 128);
  cvt(q_w2, qw2, 128, 128, 128, 128);
  cvt(k_w0, kw0, 128, 64, 128, 64);
  cvt(k_w1, kw1, 128, 128, 128, 128);
  cvt(k_w2, kw2, 128, 128, 128, 128);
  cvt(clf_w0, cw0, 1408, 2816, 1404, 2809);
  cvt(clf_w1, cw1, 704, 1408, 702, 1404);

  // Recurrent scan: 64 blocks (one per batch), 8 wave32s, big dynamic LDS.
  scan_kernel<<<64, 256, SMEM_SCAN, stream>>>(
      x, w_emb, b_emb, wih, whh, b_ih, b_hh,
      qw0, q_b0, qw1, q_b1, qw2, q_b2,
      kw0, k_b0, kw1, k_b1, kw2, k_b2,
      gate_b, pred_w, pred_b, mix, pred);

  // mean_FNC over T
  meanfnc_kernel<<<dim3(64, 11), 256, 0, stream>>>(mix, meanf);

  // Classifier: z1 = relu(mix @ W0^T + b0), z2 = relu(z1 @ W1^T + b1)
  clf_gemm_kernel<<<dim3(256, 11), 256, 0, stream>>>(
      (const void*)mix, 1, 2809, cw0, clf_b0, 1404, z1, 1408, 2816, 128);
  clf_gemm_kernel<<<dim3(256, 11), 256, 0, stream>>>(
      (const void*)z1, 0, 1408, cw1, clf_b1, 702, z2, 704, 1408, 64);

  logits_kernel<<<64, 256, 0, stream>>>(z2, clf_w2, clf_b2, logits);
  xcopy_kernel<<<(64 * 255 * 53 + 255) / 256, 256, 0, stream>>>(x, xout);
}
